// CVR_68315749810577
// MI455X (gfx1250) — matrix operand order repaired and checked by reading, not compiled, not run on hardware
//
#include <hip/hip_runtime.h>
#include <math.h>

// ---------------- model dims ----------------
constexpr int C_B = 128, C_V = 16, C_D = 2048, C_NH = 4;
constexpr int C_NHD = 8192, C_HID = 8192, C_M = 8, C_COORD = 256;
constexpr int C_KBITS = 3, C_NCLS = 100;
constexpr int C_T1 = C_B * C_V;   // 2048 tokens (mha1/ff1)
constexpr int C_T2 = C_B * C_M;   // 1024 tokens (post-OT)

// ---------------- WMMA GEMM tiling ----------------
constexpr int GBM = 128, GBN = 128, GBK = 32, LDAH = 40; // LDS row stride (halves), padded

typedef __attribute__((ext_vector_type(16))) _Float16 v16h;
typedef __attribute__((ext_vector_type(8)))  _Float16 v8h;
typedef __attribute__((ext_vector_type(8)))  float    v8f;

// load a 128x32 f32 tile into 4 float4 regs per thread (guards only in !FULL)
template<bool FULL>
__device__ __forceinline__ void ld_tile(const float* __restrict__ P, int rows, int K,
                                        int gr0, int gk0, int tid, float4* r)
{
#pragma unroll
  for (int i = 0; i < 4; ++i) {
    int fi = tid + i * 256;
    int row = fi >> 3;
    int col = (fi & 7) << 2;
    int gr = gr0 + row, gk = gk0 + col;
    float4 t = {0.f, 0.f, 0.f, 0.f};
    if (FULL) {
      const float* p = P + (size_t)gr * K + gk;
      t = *(const float4*)p;
      __builtin_prefetch(p + GBK, 0, 0);       // speculative, one tile ahead
    } else {
      if (gr < rows) {
        const float* p = P + (size_t)gr * K + gk;
        if (gk + 3 < K) t = *(const float4*)p;
        else {
          if (gk     < K) t.x = p[0];
          if (gk + 1 < K) t.y = p[1];
          if (gk + 2 < K) t.z = p[2];
          if (gk + 3 < K) t.w = p[3];
        }
      }
    }
    r[i] = t;
  }
}

// convert + store the 4 float4 regs into an LDS f16 tile
__device__ __forceinline__ void st_tile(_Float16* S, int tid, const float4* r)
{
#pragma unroll
  for (int i = 0; i < 4; ++i) {
    int fi = tid + i * 256;
    int row = fi >> 3;
    int col = (fi & 7) << 2;
    _Float16* dp = &S[row * LDAH + col];
    dp[0] = (_Float16)r[i].x; dp[1] = (_Float16)r[i].y;
    dp[2] = (_Float16)r[i].z; dp[3] = (_Float16)r[i].w;
  }
}

// OUT[m,n] = act(alpha*acc*scale[n] + shift[n]) + R[m,n]
// act: 0=none, 1=relu, 2=exp.  A:[Mr,K] f32, W:[Nc,K] f32 (row-major; computes A@W^T)
// 128x128 block tile, 8 waves (4 along M x 2 along N), wave tile 32x64 = 8 WMMA/K-step.
// Double-buffered LDS, software-pipelined global->LDS staging.
template<bool FULL>
__global__ __launch_bounds__(256) void k_gemm(
    const float* __restrict__ A, const float* __restrict__ W,
    const float* __restrict__ scale, const float* __restrict__ shift,
    const float* __restrict__ R, float* __restrict__ OUT,
    int Mr, int Nc, int K, float alpha, int act)
{
  __shared__ alignas(16) _Float16 As[2][GBM * LDAH];
  __shared__ alignas(16) _Float16 Bs[2][GBN * LDAH];
  const int tid  = threadIdx.x;
  const int lane = tid & 31;
  const int wave = tid >> 5;
  const int wm   = wave & 3;    // 4 waves along M (32 rows each)
  const int wn   = wave >> 2;   // 2 waves along N (64 cols each)
  const int bM   = blockIdx.y * GBM;
  const int bN   = blockIdx.x * GBN;
  const int l15  = lane & 15;
  const int hiH  = lane >> 4;

  v8f acc[2][4];
#pragma unroll
  for (int i = 0; i < 2; ++i)
#pragma unroll
    for (int j = 0; j < 4; ++j)
#pragma unroll
      for (int r = 0; r < 8; ++r) acc[i][j][r] = 0.0f;

  float4 ra[4], rb[4];
  ld_tile<FULL>(A, Mr, K, bM, 0, tid, ra);
  ld_tile<FULL>(W, Nc, K, bN, 0, tid, rb);
  st_tile(As[0], tid, ra);
  st_tile(Bs[0], tid, rb);
  __syncthreads();

  int buf = 0;
  for (int k0 = 0; k0 < K; k0 += GBK) {
    const bool hasNext = (k0 + GBK) < K;   // uniform
    if (hasNext) {
      ld_tile<FULL>(A, Mr, K, bM, k0 + GBK, tid, ra);   // loads in flight during WMMA
      ld_tile<FULL>(W, Nc, K, bN, k0 + GBK, tid, rb);
    }

    const _Float16* Ab = As[buf];
    const _Float16* Bb = Bs[buf];
    // A 16-bit 16x32 frag: lanes 0-15 -> K 0..7 / 16..23 ; lanes 16-31 -> K 8..15 / 24..31
    const int akb = hiH * 8;
    // B 16-bit 32x16 frag: lanes 0-15 -> K 0..7 / 16..23 ; lanes 16-31 -> K 8..15 / 24..31 (col = lane&15)
    const int bkb = hiH * 8;
    v16h af[2];
#pragma unroll
    for (int mi = 0; mi < 2; ++mi) {
      const _Float16* p = &Ab[(wm * 32 + mi * 16 + l15) * LDAH + akb];
      v8h lo = *(const v8h*)p;
      v8h hi = *(const v8h*)(p + 16);
      af[mi] = __builtin_shufflevector(lo, hi, 0,1,2,3,4,5,6,7,8,9,10,11,12,13,14,15);
    }
#pragma unroll
    for (int ni = 0; ni < 4; ++ni) {
      const _Float16* p = &Bb[(wn * 64 + ni * 16 + l15) * LDAH + bkb];
      v8h lo = *(const v8h*)p;
      v8h hi = *(const v8h*)(p + 16);
      v16h bf = __builtin_shufflevector(lo, hi, 0,1,2,3,4,5,6,7,8,9,10,11,12,13,14,15);
#pragma unroll
      for (int mi = 0; mi < 2; ++mi)
        acc[mi][ni] = __builtin_amdgcn_wmma_f32_16x16x32_f16(
            false, af[mi], false, bf, (short)0, acc[mi][ni], false, false);
    }

    if (hasNext) {
      st_tile(As[buf ^ 1], tid, ra);
      st_tile(Bs[buf ^ 1], tid, rb);
    }
    __syncthreads();
    buf ^= 1;
  }

  // ---- epilogue: C layout = 8 VGPRs, lanes 0-15: M=r, lanes 16-31: M=8+r, N=lane&15 ----
  const int rbase = hiH * 8;
#pragma unroll
  for (int mi = 0; mi < 2; ++mi) {
#pragma unroll
    for (int ni = 0; ni < 4; ++ni) {
      int n = bN + wn * 64 + ni * 16 + l15;
      if (!FULL && n >= Nc) continue;
      float sc = scale ? scale[n] : 1.0f;
      float sh = shift ? shift[n] : 0.0f;
#pragma unroll
      for (int r = 0; r < 8; ++r) {
        int m = bM + wm * 32 + mi * 16 + rbase + r;
        if (!FULL && m >= Mr) continue;
        float v = acc[mi][ni][r] * alpha * sc + sh;
        if (act == 1) v = fmaxf(v, 0.0f);
        else if (act == 2) v = __expf(v);
        if (R) v += R[(size_t)m * Nc + n];
        OUT[(size_t)m * Nc + n] = v;
      }
    }
  }
}

// scores + softmax: one block per (b,h,q); S keys (S<=16), head dim C_D
__global__ void k_attn_scores(const float* __restrict__ Q, const float* __restrict__ Kh,
                              float* __restrict__ att, int S, float rscale)
{
  int bi = blockIdx.x;                 // bb*NH*S + h*S + q
  int q  = bi % S;
  int h  = (bi / S) % C_NH;
  int bb = bi / (S * C_NH);
  int tid = threadIdx.x;
  __shared__ float red[256];
  __shared__ float srow[16];
  const float* qrow = Q + (size_t)(bb * S + q) * C_NHD + (size_t)h * C_D;
  int g = tid >> 4, t2 = tid & 15;
  float partial = 0.0f;
  if (g < S) {
    const float* krow = Kh + (size_t)(bb * S + g) * C_NHD + (size_t)h * C_D;
    for (int d = t2; d < C_D; d += 16) partial += qrow[d] * krow[d];
  }
  red[tid] = partial;
  __syncthreads();
  if (t2 == 0 && g < S) {
    float s = 0.f;
    for (int i = 0; i < 16; ++i) s += red[g * 16 + i];
    srow[g] = s * rscale;
  }
  __syncthreads();
  if (tid == 0) {
    float mx = -1e30f;
    for (int k = 0; k < S; ++k) mx = fmaxf(mx, srow[k]);
    float se = 0.f;
    for (int k = 0; k < S; ++k) { float e = __expf(srow[k] - mx); srow[k] = e; se += e; }
    float inv = 1.0f / se;
    float* arow = att + (size_t)bi * S;
    for (int k = 0; k < S; ++k) arow[k] = srow[k] * inv;
  }
}

// O[token, col] = sum_k att[b,h,q,k] * Vh[(b*S+k), col]   (col = h*C_D + d)
__global__ void k_attn_av(const float* __restrict__ att, const float* __restrict__ Vh,
                          float* __restrict__ O, int S)
{
  size_t idx = (size_t)blockIdx.x * 256 + threadIdx.x;
  int col = (int)(idx % C_NHD);
  int token = (int)(idx / C_NHD);
  int h = col >> 11;                   // / C_D
  int q = token % S, bb = token / S;
  const float* arow = att + ((size_t)(bb * C_NH + h) * S + q) * S;
  float s = 0.f;
  for (int k = 0; k < S; ++k)
    s += arow[k] * Vh[(size_t)(bb * S + k) * C_NHD + col];
  O[idx] = s;
}

// LayerNorm over D=2048, one row per block
__global__ void k_layernorm(const float* __restrict__ x, const float* __restrict__ g,
                            const float* __restrict__ b, float* __restrict__ out)
{
  int row = blockIdx.x, tid = threadIdx.x;
  const float* xr = x + (size_t)row * C_D;
  float lv[8], s = 0.f;
#pragma unroll
  for (int i = 0; i < 8; ++i) { lv[i] = xr[tid + i * 256]; s += lv[i]; }
  __shared__ float red[256];
  red[tid] = s; __syncthreads();
  for (int off = 128; off > 0; off >>= 1) { if (tid < off) red[tid] += red[tid + off]; __syncthreads(); }
  float mean = red[0] * (1.0f / C_D);
  __syncthreads();
  s = 0.f;
#pragma unroll
  for (int i = 0; i < 8; ++i) { float d = lv[i] - mean; s += d * d; }
  red[tid] = s; __syncthreads();
  for (int off = 128; off > 0; off >>= 1) { if (tid < off) red[tid] += red[tid + off]; __syncthreads(); }
  float rstd = rsqrtf(red[0] * (1.0f / C_D) + 1e-5f);
  float* orow = out + (size_t)row * C_D;
#pragma unroll
  for (int i = 0; i < 8; ++i) { int d = tid + i * 256; orow[d] = (lv[i] - mean) * rstd * g[d] + b[d]; }
}

// row-wise l2 normalize: out = x / max(||x||, eps)
__global__ void k_l2norm(const float* __restrict__ x, float* __restrict__ out, int Dd, float eps)
{
  int row = blockIdx.x, tid = threadIdx.x;
  const float* xr = x + (size_t)row * Dd;
  float s = 0.f;
  for (int d = tid; d < Dd; d += 256) { float v = xr[d]; s += v * v; }
  __shared__ float red[256];
  red[tid] = s; __syncthreads();
  for (int off = 128; off > 0; off >>= 1) { if (tid < off) red[tid] += red[tid + off]; __syncthreads(); }
  float inv = 1.0f / fmaxf(sqrtf(red[0]), eps);
  float* orow = out + (size_t)row * Dd;
  for (int d = tid; d < Dd; d += 256) orow[d] = xr[d] * inv;
}

// Sinkhorn: per-batch block (128 thr); K is [S, Ml] in LDS
__global__ void k_sinkhorn(const float* __restrict__ Km, float* __restrict__ T,
                           int S, int Ml, int iters, float a)
{
  int b = blockIdx.x, tid = threadIdx.x;
  __shared__ float Ks[C_V * C_M];
  __shared__ float u[C_V];
  __shared__ float v[C_M];
  if (tid < S * Ml) Ks[tid] = Km[(size_t)b * S * Ml + tid];
  if (tid < S)  u[tid] = 1.0f;
  if (tid < Ml) v[tid] = 1.0f;
  __syncthreads();
  for (int it = 0; it < iters; ++it) {
    if (tid < S) { float s = 0.f; for (int m = 0; m < Ml; ++m) s += Ks[tid * Ml + m] * v[m]; u[tid] = a / s; }
    __syncthreads();
    if (tid < Ml) { float s = 0.f; for (int ss = 0; ss < S; ++ss) s += u[ss] * Ks[ss * Ml + tid]; v[tid] = 1.0f / s; }
    __syncthreads();
  }
  if (tid < S * Ml) { int ss = tid / Ml, m = tid % Ml; T[(size_t)b * S * Ml + tid] = u[ss] * Ks[tid] * v[m]; }
}

// out1[b,m,d] = sum_s T[b,s,m] * X[b*V+s, d]
__global__ void k_otk_pool(const float* __restrict__ T, const float* __restrict__ X,
                           float* __restrict__ out1)
{
  size_t idx = (size_t)blockIdx.x * 256 + threadIdx.x;   // B*M*D
  int d = (int)(idx % C_D);
  int m = (int)((idx / C_D) % C_M);
  int b = (int)(idx / ((size_t)C_D * C_M));
  float s = 0.f;
  for (int ss = 0; ss < C_V; ++ss)
    s += T[((size_t)b * C_V + ss) * C_M + m] * X[((size_t)b * C_V + ss) * C_D + d];
  out1[idx] = s;
}

// fold eval-mode BN into per-channel affine: BN(acc+bias) = acc*scale + shift
__global__ void k_bnfold(const float* g, const float* b, const float* rm, const float* rv,
                         const float* bias, float* scale, float* shift, int n)
{
  int i = blockIdx.x * 256 + threadIdx.x;
  if (i < n) {
    float s = g[i] * rsqrtf(rv[i] + 1e-5f);
    scale[i] = s;
    shift[i] = b[i] + (bias[i] - rm[i]) * s;
  }
}

__global__ void k_meanpool(const float* __restrict__ x, float* __restrict__ out)
{
  size_t idx = (size_t)blockIdx.x * 256 + threadIdx.x;   // B*D
  int d = (int)(idx % C_D), b = (int)(idx / C_D);
  float s = 0.f;
  for (int m = 0; m < C_M; ++m) s += x[((size_t)b * C_M + m) * C_D + d];
  out[idx] = s * (1.0f / C_M);
}

extern "C" void kernel_launch(void* const* d_in, const int* in_sizes, int n_in,
                              void* d_out, int out_size, void* d_ws, size_t ws_size,
                              hipStream_t stream)
{
  (void)in_sizes; (void)n_in; (void)out_size; (void)ws_size;
  const float* x     = (const float*)d_in[0];
  // params flattened in setup_inputs() insertion order (after x, num_views, batch_size, max_num_views)
  const float* m1_wq = (const float*)d_in[4];  const float* m1_bq = (const float*)d_in[5];
  const float* m1_wk = (const float*)d_in[6];  const float* m1_bk = (const float*)d_in[7];
  const float* m1_wv = (const float*)d_in[8];  const float* m1_bv = (const float*)d_in[9];
  const float* m1_wo = (const float*)d_in[10]; const float* m1_bo = (const float*)d_in[11];
  const float* ln1_g = (const float*)d_in[12]; const float* ln1_b = (const float*)d_in[13];
  const float* f1_w1 = (const float*)d_in[14]; const float* f1_b1 = (const float*)d_in[15];
  const float* f1_w2 = (const float*)d_in[16]; const float* f1_b2 = (const float*)d_in[17];
  const float* otkw  = (const float*)d_in[18];
  const float* f2_w1 = (const float*)d_in[19]; const float* f2_b1 = (const float*)d_in[20];
  const float* f2_w2 = (const float*)d_in[21]; const float* f2_b2 = (const float*)d_in[22];
  const float* f2_g  = (const float*)d_in[23]; const float* f2_b  = (const float*)d_in[24];
  const float* f2_rm = (const float*)d_in[25]; const float* f2_rv = (const float*)d_in[26];
  const float* ce_w1 = (const float*)d_in[27]; const float* ce_b1 = (const float*)d_in[28];
  const float* ce_w2 = (const float*)d_in[29]; const float* ce_b2 = (const float*)d_in[30];
  const float* cd_w1 = (const float*)d_in[31]; const float* cd_b1 = (const float*)d_in[32];
  const float* cd_w2 = (const float*)d_in[33]; const float* cd_b2 = (const float*)d_in[34];
  const float* m2_wq = (const float*)d_in[35]; const float* m2_bq = (const float*)d_in[36];
  const float* m2_wk = (const float*)d_in[37]; const float* m2_bk = (const float*)d_in[38];
  const float* m2_wv = (const float*)d_in[39]; const float* m2_bv = (const float*)d_in[40];
  const float* m2_wo = (const float*)d_in[41]; const float* m2_bo = (const float*)d_in[42];
  const float* f3_w1 = (const float*)d_in[43]; const float* f3_b1 = (const float*)d_in[44];
  const float* f3_w2 = (const float*)d_in[45]; const float* f3_b2 = (const float*)d_in[46];
  const float* f3_g  = (const float*)d_in[47]; const float* f3_b  = (const float*)d_in[48];
  const float* f3_rm = (const float*)d_in[49]; const float* f3_rv = (const float*)d_in[50];
  const float* cls_w = (const float*)d_in[51]; const float* cls_b = (const float*)d_in[52];

  // ---- workspace layout (floats), with reuse ----
  float* ws = (float*)d_ws;
  const size_t BIG = (size_t)C_T1 * C_NHD;       // 16,777,216
  float* bigA = ws;                              // qh1 -> o1 -> qh2 -> o2
  float* bigB = ws + BIG;                        // kh1 -> ffn hidden -> kh2 -> ffn hidden
  float* bigC = ws + 2 * BIG;                    // vh1 -> vh2
  float* mid0 = ws + 3 * BIG;                    // attnout -> xn -> out2   [T1,D]
  float* mid1 = mid0 + (size_t)C_T1 * C_D;       // out0 -> q               [T1,D]
  float* sm   = mid1 + (size_t)C_T1 * C_D;
  float* att1 = sm;                              // 131072 (also att2)
  float* wn   = att1 + 131072;                   // 16384
  float* Km   = wn   + 16384;                    // 16384
  float* Tm   = Km   + 16384;                    // 16384
  float* out1 = Tm   + 16384;                    // 2,097,152
  float* ceh  = out1 + (size_t)C_T2 * C_D;       // 262144 (ce & cd hidden)
  float* pos0 = ceh  + (size_t)C_T2 * C_COORD;   // 4096
  float* sc2  = pos0 + 4096;                     // 8192
  float* sh2  = sc2  + C_HID;
  float* sc3  = sh2  + C_HID;
  float* sh3  = sc3  + C_HID;

  float* logits = (float*)d_out;
  float* pooled = logits + (size_t)C_B * C_NCLS;

  dim3 blk(256);
  auto GEMM = [&](const float* A, const float* W, const float* scale, const float* shift,
                  const float* R, float* OUT, int M_, int N_, int K_, float alpha, int act) {
    dim3 grid((unsigned)((N_ + GBN - 1) / GBN), (unsigned)((M_ + GBM - 1) / GBM));
    const bool full = (M_ % GBM == 0) && (N_ % GBN == 0) && (K_ % GBK == 0);
    if (full)
      k_gemm<true><<<grid, blk, 0, stream>>>(A, W, scale, shift, R, OUT, M_, N_, K_, alpha, act);
    else
      k_gemm<false><<<grid, blk, 0, stream>>>(A, W, scale, shift, R, OUT, M_, N_, K_, alpha, act);
  };
  const float rscale = 0.022097086912079608f;    // 1/sqrt(2048)

  // ---- MHA1 ----
  GEMM(x, m1_wq, nullptr, m1_bq, nullptr, bigA, C_T1, C_NHD, C_D, 1.f, 0);
  GEMM(x, m1_wk, nullptr, m1_bk, nullptr, bigB, C_T1, C_NHD, C_D, 1.f, 0);
  GEMM(x, m1_wv, nullptr, m1_bv, nullptr, bigC, C_T1, C_NHD, C_D, 1.f, 0);
  k_attn_scores<<<C_B * C_NH * C_V, blk, 0, stream>>>(bigA, bigB, att1, C_V, rscale);
  k_attn_av<<<(unsigned)(((size_t)C_T1 * C_NHD) / 256), blk, 0, stream>>>(att1, bigC, bigA, C_V);
  GEMM(bigA, m1_wo, nullptr, m1_bo, x, mid0, C_T1, C_D, C_NHD, 1.f, 0);     // + residual z
  k_layernorm<<<C_T1, blk, 0, stream>>>(mid0, ln1_g, ln1_b, mid1);          // out0
  // ---- FF1 (residual) ----
  GEMM(mid1, f1_w1, nullptr, f1_b1, nullptr, bigB, C_T1, C_HID, C_D, 1.f, 1);
  GEMM(bigB, f1_w2, nullptr, f1_b2, mid1, mid1, C_T1, C_D, C_HID, 1.f, 0);
  // ---- OT pooling ----
  k_l2norm<<<C_T1, blk, 0, stream>>>(mid1, mid0, C_D, 1e-6f);
  k_l2norm<<<C_M, blk, 0, stream>>>(otkw, wn, C_D, 1e-6f);
  GEMM(mid0, wn, nullptr, nullptr, nullptr, Km, C_T1, C_M, C_D, 10.0f, 2);  // exp(cos/eps)
  k_sinkhorn<<<C_B, dim3(128), 0, stream>>>(Km, Tm, C_V, C_M, 10, 0.5f);    // a = M/S
  k_otk_pool<<<(unsigned)(((size_t)C_B * C_M * C_D) / 256), blk, 0, stream>>>(Tm, mid1, out1);
  // ---- FF2 (BN, residual) ----
  k_bnfold<<<C_HID / 256, blk, 0, stream>>>(f2_g, f2_b, f2_rm, f2_rv, f2_b1, sc2, sh2, C_HID);
  GEMM(out1, f2_w1, sc2, sh2, nullptr, bigB, C_T2, C_HID, C_D, 1.f, 1);
  GEMM(bigB, f2_w2, nullptr, f2_b2, out1, out1, C_T2, C_D, C_HID, 1.f, 0);
  // ---- coordinate bottleneck ----
  GEMM(out1, ce_w1, nullptr, ce_b1, nullptr, ceh, C_T2, C_COORD, C_D, 1.f, 1);
  GEMM(ceh, ce_w2, nullptr, ce_b2, nullptr, pos0, C_T2, C_KBITS, C_COORD, 1.f, 0);
  k_l2norm<<<C_T2, blk, 0, stream>>>(pos0, pos0, C_KBITS, 1e-6f);
  GEMM(pos0, cd_w1, nullptr, cd_b1, nullptr, ceh, C_T2, C_COORD, C_KBITS, 1.f, 1);
  GEMM(ceh, cd_w2, nullptr, cd_b2, out1, mid1, C_T2, C_D, C_COORD, 1.f, 0); // q = out1 + pos1
  // ---- MHA2 (q,q,out1) ----
  GEMM(mid1, m2_wq, nullptr, m2_bq, nullptr, bigA, C_T2, C_NHD, C_D, 1.f, 0);
  GEMM(mid1, m2_wk, nullptr, m2_bk, nullptr, bigB, C_T2, C_NHD, C_D, 1.f, 0);
  GEMM(out1, m2_wv, nullptr, m2_bv, nullptr, bigC, C_T2, C_NHD, C_D, 1.f, 0);
  k_attn_scores<<<C_B * C_NH * C_M, blk, 0, stream>>>(bigA, bigB, att1, C_M, rscale);
  k_attn_av<<<(unsigned)(((size_t)C_T2 * C_NHD) / 256), blk, 0, stream>>>(att1, bigC, bigA, C_M);
  GEMM(bigA, m2_wo, nullptr, m2_bo, mid1, mid0, C_T2, C_D, C_NHD, 1.f, 0);  // out2 = q + proj
  // ---- FF3 (BN, residual) ----
  k_bnfold<<<C_HID / 256, blk, 0, stream>>>(f3_g, f3_b, f3_rm, f3_rv, f3_b1, sc3, sh3, C_HID);
  GEMM(mid0, f3_w1, sc3, sh3, nullptr, bigB, C_T2, C_HID, C_D, 1.f, 1);
  GEMM(bigB, f3_w2, nullptr, f3_b2, mid0, mid0, C_T2, C_D, C_HID, 1.f, 0);
  // ---- pool + classifier ----
  k_meanpool<<<(unsigned)(((size_t)C_B * C_D) / 256), blk, 0, stream>>>(mid0, pooled);
  GEMM(pooled, cls_w, nullptr, cls_b, nullptr, logits, C_B, C_NCLS, C_D, 1.f, 0);
}
